// GroupConvAttention_68066641707315
// MI455X (gfx1250) — matrix operand order repaired and checked by reading, not compiled, not run on hardware
//
#include <hip/hip_runtime.h>
#include <hip/hip_bf16.h>

// ---------------------------------------------------------------------------
// Problem constants (B,C,H,W)=(4,256,64,64), hw=1024 -> hs=ws=32, rate=2,
// vk=4, pad=1.  Per batch GEMM: M=1024 (QR), K=1024 (mn), N=4096 (uv*256+c).
// ---------------------------------------------------------------------------
#define GB  4
#define GC  256
#define GH  64
#define GW  64
#define GM  1024            // QR
#define GK  1024            // mn
#define GN  4096            // (u*4+v)*256 + c

typedef __attribute__((ext_vector_type(16))) __bf16 v16bf;
typedef __attribute__((ext_vector_type(8)))  __bf16 v8bf;
typedef __attribute__((ext_vector_type(8)))  float  v8f;

// ---------------------------------------------------------------------------
// Prep 1: scores (f32) -> bf16, identical layout [b][qr][mn] (A operand).
// ---------------------------------------------------------------------------
__global__ __launch_bounds__(256) void prep_scores_kernel(
    const float* __restrict__ s, __bf16* __restrict__ o)
{
    int t = blockIdx.x * 256 + threadIdx.x;        // GB*GM*GK / 4 threads, x4
    int i = t * 4;
    o[i + 0] = (__bf16)s[i + 0];
    o[i + 1] = (__bf16)s[i + 1];
    o[i + 2] = (__bf16)s[i + 2];
    o[i + 3] = (__bf16)s[i + 3];
}

// ---------------------------------------------------------------------------
// Prep 2: build V_t[b][ncol][k] in bf16, ncol = (u*4+v)*256 + c, k = m*32+n.
// V_t[b][ncol][k] = x_pad[b,c,2m+u,2n+v] = ori_x edge-clamped at (2m+u-1,
// 2n+v-1).  N-major / K-contiguous == B^T, so B fragments are contiguous.
// ---------------------------------------------------------------------------
__global__ __launch_bounds__(256) void prep_value_kernel(
    const float* __restrict__ x, __bf16* __restrict__ vt)
{
    int t  = blockIdx.x * 256 + threadIdx.x;       // GB*GN*GK threads
    int k  = t & (GK - 1);
    int nc = (t >> 10) & (GN - 1);
    int b  = t >> 22;
    int m  = k >> 5, nn = k & 31;
    int uv = nc >> 8, c = nc & 255;
    int u  = uv >> 2, v = uv & 3;
    int h  = 2 * m + u - 1;  h = min(max(h, 0), GH - 1);
    int w  = 2 * nn + v - 1; w = min(max(w, 0), GW - 1);
    vt[t] = (__bf16)x[(((b * GC + c) * GH) + h) * GW + w];
}

// ---------------------------------------------------------------------------
// GEMM (per batch): P[M][N] = S[M][K] @ V_t^T, bf16 WMMA, f32 accumulate.
// 256 threads = 8 waves, workgroup tile 128x128, wave tile 64x32 (4x2 WMMA).
// Fragments loaded straight from global (operands live in 192MB L2).
//
// CDNA5 16-bit A layout (ISA 7.12.2): lane (r, half): row M=r, K chunks
// [half*8, half*8+8) and [16+half*8, ...).  B (K x 16): lane (r, half):
// col N=r, same K chunks as A -> two 16B loads from B^T.
// C/D: VGPR i -> (M = half*8 + i, N = r).
//
// K loop is a ping-pong unroll-by-2: two fixed fragment register sets
// alternate roles, so there are no cross-iteration register copies (the
// copy-based pipeline lowered to v_dual_mov storms + WMMA WAR v_nops).
// ---------------------------------------------------------------------------
__device__ __forceinline__ v16bf load_a_frag(const __bf16* p)
{
    v8bf lo = *(const v8bf*)(p);
    v8bf hi = *(const v8bf*)(p + 16);
    return __builtin_shufflevector(lo, hi, 0, 1, 2, 3, 4, 5, 6, 7,
                                   8, 9, 10, 11, 12, 13, 14, 15);
}

__device__ __forceinline__ void load_frags(const __bf16* Ap, const __bf16* Bp,
                                           int k0, v16bf a[4], v16bf bb[2])
{
#pragma unroll
    for (int im = 0; im < 4; ++im) a[im] = load_a_frag(Ap + im * 16 * GK + k0);
#pragma unroll
    for (int in = 0; in < 2; ++in) bb[in] = load_a_frag(Bp + in * 16 * GK + k0);
}

__device__ __forceinline__ void do_wmma(const v16bf a[4], const v16bf bb[2],
                                        v8f acc[4][2])
{
#pragma unroll
    for (int im = 0; im < 4; ++im)
#pragma unroll
        for (int in = 0; in < 2; ++in)
            acc[im][in] = __builtin_amdgcn_wmma_f32_16x16x32_bf16(
                false, a[im], false, bb[in], (short)0, acc[im][in],
                false, false);
}

__global__ __launch_bounds__(256) void wmma_gemm_kernel(
    const __bf16* __restrict__ S,   // [GM][GK]  (one batch)
    const __bf16* __restrict__ Vt,  // [GN][GK]  (one batch)
    float* __restrict__ P)          // [GM][GN]
{
    const int tid  = threadIdx.x;
    const int wv   = tid >> 5;
    const int lane = tid & 31;
    const int half = lane >> 4;
    const int r    = lane & 15;
    const int wm   = wv >> 2;                     // 0..1
    const int wn   = wv & 3;                      // 0..3
    const int row0 = blockIdx.y * 128 + wm * 64;  // wave M base
    const int col0 = blockIdx.x * 128 + wn * 32;  // wave N base

    const __bf16* Ap = S  + (row0 + r) * GK + half * 8;
    const __bf16* Bp = Vt + (col0 + r) * GK + half * 8;

    v8f acc[4][2] = {};
    v16bf a0[4], b0[2], a1[4], b1[2];

    load_frags(Ap, Bp, 0, a0, b0);
    for (int k0 = 0; k0 < GK - 64; k0 += 64) {
        load_frags(Ap, Bp, k0 + 32, a1, b1);   // prefetch odd step
        do_wmma(a0, b0, acc);                  // compute even step
        load_frags(Ap, Bp, k0 + 64, a0, b0);   // prefetch next even step
        do_wmma(a1, b1, acc);                  // compute odd step
    }
    load_frags(Ap, Bp, GK - 32, a1, b1);
    do_wmma(a0, b0, acc);
    do_wmma(a1, b1, acc);

    // Store D: row = row0 + 16*im + half*8 + i, col = col0 + 16*in + r.
#pragma unroll
    for (int im = 0; im < 4; ++im)
#pragma unroll
        for (int in = 0; in < 2; ++in) {
            float* pd = P + (size_t)(row0 + 16 * im + half * 8) * GN
                          + (col0 + 16 * in + r);
#pragma unroll
            for (int i = 0; i < 8; ++i)
                pd[(size_t)i * GN] = acc[im][in][i];
        }
}

// ---------------------------------------------------------------------------
// Epilogue (per batch): overlap-add of the 4x4 stride-2 patches, /4, *alpha,
// + ori_x.  Output pixel (h,w) sums P over {(Q,u): 2Q+u = h+1, 0<=Q<32} x
// {(R,v): 2R+v = w+1, 0<=R<32} (<=4 terms).  Deterministic gather, no atomics.
// Thread map: t = (h*64 + w)*256 + c  -> contiguous P reads along c per wave.
// ---------------------------------------------------------------------------
__global__ __launch_bounds__(256) void epilogue_kernel(
    const float* __restrict__ x, const float* __restrict__ P,
    const float* __restrict__ alpha, float* __restrict__ out, int b)
{
    int t = blockIdx.x * 256 + threadIdx.x;       // GC*GH*GW threads
    int c = t & 255;
    int w = (t >> 8) & 63;
    int h = t >> 14;

    const int hp = h + 1, wp = w + 1;
    const int u0 = hp & 1, v0 = wp & 1;
    float sum = 0.0f;
#pragma unroll
    for (int du = 0; du < 2; ++du) {
        int u = u0 + 2 * du;
        int Q = (hp - u) >> 1;
        if ((unsigned)Q < 32u) {
#pragma unroll
            for (int dv = 0; dv < 2; ++dv) {
                int v = v0 + 2 * dv;
                int R = (wp - v) >> 1;
                if ((unsigned)R < 32u)
                    sum += P[(size_t)(Q * 32 + R) * GN + (u * 4 + v) * 256 + c];
            }
        }
    }
    size_t oi = (((size_t)b * GC + c) * GH + h) * GW + w;
    out[oi] = x[oi] + sum * (0.25f * alpha[0]);
}

// ---------------------------------------------------------------------------
// Launch: prep (all batches) -> per batch: GEMM, epilogue.  Single in-order
// stream provides all ordering.  Workspace: S_bf 8MB | V_t 33.5MB | P 16.8MB.
// ---------------------------------------------------------------------------
extern "C" void kernel_launch(void* const* d_in, const int* in_sizes, int n_in,
                              void* d_out, int out_size, void* d_ws, size_t ws_size,
                              hipStream_t stream)
{
    const float* ori_x  = (const float*)d_in[0];
    const float* scores = (const float*)d_in[1];
    const float* alpha  = (const float*)d_in[2];
    float* out = (float*)d_out;

    char* ws = (char*)d_ws;
    const size_t S_BYTES = (size_t)GB * GM * GK * sizeof(__bf16);   //  8 MiB
    const size_t V_BYTES = (size_t)GB * GN * GK * sizeof(__bf16);   // 32 MiB
    __bf16* S_bf = (__bf16*)ws;
    __bf16* V_t  = (__bf16*)(ws + S_BYTES);
    float*  P    = (float*)(ws + S_BYTES + V_BYTES);                // 16 MiB

    // Stage 1: operand prep (bf16 conversion + value-gather into B^T layout).
    {
        int n4 = GB * GM * GK / 4;
        prep_scores_kernel<<<n4 / 256, 256, 0, stream>>>(scores, S_bf);
        int nv = GB * GN * GK;
        prep_value_kernel<<<nv / 256, 256, 0, stream>>>(ori_x, V_t);
    }

    // Stage 2+3: per-batch WMMA GEMM then deterministic overlap-add epilogue.
    dim3 ggrid(GN / 128, GM / 128, 1);   // (32, 8)
    int  ne = GC * GH * GW;              // 1,048,576
    for (int b = 0; b < GB; ++b) {
        wmma_gemm_kernel<<<ggrid, 256, 0, stream>>>(
            S_bf + (size_t)b * GM * GK, V_t + (size_t)b * GN * GK, P);
        epilogue_kernel<<<ne / 256, 256, 0, stream>>>(ori_x, P, alpha, out, b);
    }
}
